// SelfAttention_61933428408352
// MI455X (gfx1250) — matrix operand order repaired and checked by reading, not compiled, not run on hardware
//
#include <hip/hip_runtime.h>
#include <hip/hip_bf16.h>

#define HID   1024
#define NH    16
#define HD    64
#define SEQ   1024
#define BATCH 4
#define ROWS  (BATCH * SEQ)   // 4096

typedef __attribute__((ext_vector_type(16))) __bf16 v16bf;
typedef __attribute__((ext_vector_type(8)))  float  v8f;
typedef __attribute__((ext_vector_type(4)))  int    v4i;
typedef __attribute__((ext_vector_type(4)))  float  v4f;

union FragAB {
    v16bf v;
    v4i   q[2];
    unsigned short u[16];
};

__device__ __forceinline__ unsigned short f32_bf16(float f) {
    unsigned int u = __float_as_uint(f);
    u += 0x7fffu + ((u >> 16) & 1u);   // round-to-nearest-even
    return (unsigned short)(u >> 16);
}

__device__ __forceinline__ v8f wmma_bf16(const FragAB& a, const FragAB& b, v8f c) {
    return __builtin_amdgcn_wmma_f32_16x16x32_bf16(
        /*neg_a=*/false, a.v, /*neg_b=*/false, b.v,
        /*c_mod=*/(short)0, c, /*reuse_a=*/false, /*reuse_b=*/false);
}

__device__ __forceinline__ float redmax16(float v) {
    #pragma unroll
    for (int off = 1; off < 16; off <<= 1) v = fmaxf(v, __shfl_xor(v, off, 32));
    return v;
}
__device__ __forceinline__ float redsum16(float v) {
    #pragma unroll
    for (int off = 1; off < 16; off <<= 1) v += __shfl_xor(v, off, 32);
    return v;
}

// ---------------------------------------------------------------------------
// 1) LayerNorm (f32 in) -> bf16 normed. One 256-thread block per row.
// ---------------------------------------------------------------------------
__global__ void ln_bf16_kernel(const float* __restrict__ x,
                               const float* __restrict__ gamma,
                               const float* __restrict__ beta,
                               unsigned short* __restrict__ out) {
    const int row = blockIdx.x;
    const int tid = threadIdx.x;              // 256 threads, 4 elems each
    const float* xr = x + (size_t)row * HID;
    v4f xv = ((const v4f*)xr)[tid];
    float s  = xv.x + xv.y + xv.z + xv.w;
    float sq = xv.x * xv.x + xv.y * xv.y + xv.z * xv.z + xv.w * xv.w;

    __shared__ float sh_s[256];
    __shared__ float sh_q[256];
    sh_s[tid] = s; sh_q[tid] = sq;
    __syncthreads();
    #pragma unroll
    for (int st = 128; st > 0; st >>= 1) {
        if (tid < st) { sh_s[tid] += sh_s[tid + st]; sh_q[tid] += sh_q[tid + st]; }
        __syncthreads();
    }
    const float mean = sh_s[0] * (1.0f / HID);
    const float var  = sh_q[0] * (1.0f / HID) - mean * mean;
    const float rstd = rsqrtf(var + 1e-5f);

    v4f g = ((const v4f*)gamma)[tid];
    v4f b = ((const v4f*)beta)[tid];
    unsigned short* orow = out + (size_t)row * HID + tid * 4;
    orow[0] = f32_bf16((xv.x - mean) * rstd * g.x + b.x);
    orow[1] = f32_bf16((xv.y - mean) * rstd * g.y + b.y);
    orow[2] = f32_bf16((xv.z - mean) * rstd * g.z + b.z);
    orow[3] = f32_bf16((xv.w - mean) * rstd * g.w + b.w);
}

// ---------------------------------------------------------------------------
// 2) f32 -> bf16 bulk conversion (weights)
// ---------------------------------------------------------------------------
__global__ void f32_to_bf16_kernel(const float* __restrict__ src,
                                   unsigned short* __restrict__ dst, int n) {
    int i = blockIdx.x * blockDim.x + threadIdx.x;
    if (i < n) dst[i] = f32_bf16(src[i]);
}

// ---------------------------------------------------------------------------
// 3) Fused QKV projection: Y = Xn @ W^T  (bf16 WMMA, f32 accum)
//    Q,K stored [B,NH,S,64]; V stored transposed [B,NH,64,S].
//    32x64 output tile per wave: B fragments reused across two M-subtiles.
//    grid = (ROWS/32, HID/64, 3), block = 32
// ---------------------------------------------------------------------------
__global__ void __launch_bounds__(32)
qkv_gemm_kernel(const unsigned short* __restrict__ Xn,
                const unsigned short* __restrict__ Wq,
                const unsigned short* __restrict__ Wk,
                const unsigned short* __restrict__ Wv,
                unsigned short* __restrict__ Q,
                unsigned short* __restrict__ K,
                unsigned short* __restrict__ Vt) {
    const int lane = threadIdx.x;
    const int half = lane >> 4;
    const int nn   = lane & 15;
    const int m0   = blockIdx.x * 32;
    const int n0   = blockIdx.y * 64;
    const int which = blockIdx.z;
    const unsigned short* W = (which == 0) ? Wq : ((which == 1) ? Wk : Wv);

    v8f acc[2][4] = {};
    const unsigned short* arow0 = Xn + (size_t)(m0 + nn) * HID;
    const unsigned short* arow1 = Xn + (size_t)(m0 + 16 + nn) * HID;

    for (int k0 = 0; k0 < HID; k0 += 32) {
        FragAB a0, a1;
        a0.q[0] = *(const v4i*)(arow0 + k0 + 16 * half);
        a0.q[1] = *(const v4i*)(arow0 + k0 + 16 * half + 8);
        a1.q[0] = *(const v4i*)(arow1 + k0 + 16 * half);
        a1.q[1] = *(const v4i*)(arow1 + k0 + 16 * half + 8);
        #pragma unroll
        for (int nt = 0; nt < 4; ++nt) {
            FragAB b;
            const unsigned short* wr =
                W + (size_t)(n0 + nt * 16 + nn) * HID + k0 + 16 * half;
            b.q[0] = *(const v4i*)(wr);
            b.q[1] = *(const v4i*)(wr + 8);
            acc[0][nt] = wmma_bf16(a0, b, acc[0][nt]);
            acc[1][nt] = wmma_bf16(a1, b, acc[1][nt]);
        }
    }

    #pragma unroll
    for (int mi = 0; mi < 2; ++mi) {
        #pragma unroll
        for (int nt = 0; nt < 4; ++nt) {
            #pragma unroll
            for (int r = 0; r < 8; ++r) {
                const int m = m0 + mi * 16 + r + 8 * half;  // row in [0, 4096)
                const int n = n0 + nt * 16 + nn;            // col in [0, 1024)
                const int bb = m >> 10, ss = m & 1023;
                const int hh = n >> 6,  dd = n & 63;
                const unsigned short val = f32_bf16(acc[mi][nt][r]);
                if (which == 0)
                    Q[(size_t)((bb * NH + hh) * SEQ + ss) * HD + dd] = val;
                else if (which == 1)
                    K[(size_t)((bb * NH + hh) * SEQ + ss) * HD + dd] = val;
                else
                    Vt[(size_t)((bb * NH + hh) * HD + dd) * SEQ + ss] = val;
            }
        }
    }
}

// ---------------------------------------------------------------------------
// 4) Flash attention. One wave per (bh, 16-query tile). Online softmax.
//    grid = (SEQ/16, BATCH*NH), block = 32
// ---------------------------------------------------------------------------
__global__ void __launch_bounds__(32)
attn_kernel(const unsigned short* __restrict__ Q,
            const unsigned short* __restrict__ K,
            const unsigned short* __restrict__ Vt,
            const float* __restrict__ mask,
            unsigned short* __restrict__ Ctx) {
    const int lane = threadIdx.x;
    const int half = lane >> 4;
    const int nn   = lane & 15;
    const int q0   = blockIdx.x * 16;
    const int bh   = blockIdx.y;
    const int b    = bh >> 4;
    const int h    = bh & 15;

    const unsigned short* Qb = Q  + (size_t)bh * SEQ * HD;
    const unsigned short* Kb = K  + (size_t)bh * SEQ * HD;
    const unsigned short* Vb = Vt + (size_t)bh * HD * SEQ;
    const float*          Mb = mask + (size_t)b * SEQ * SEQ;

    // Q A-fragments for d-slices [0,32) and [32,64)
    FragAB aQ[2];
    const unsigned short* qrow = Qb + (size_t)(q0 + nn) * HD;
    #pragma unroll
    for (int t = 0; t < 2; ++t) {
        aQ[t].q[0] = *(const v4i*)(qrow + t * 32 + 16 * half);
        aQ[t].q[1] = *(const v4i*)(qrow + t * 32 + 16 * half + 8);
    }

    float m_run[8], l_run[8];
    #pragma unroll
    for (int r = 0; r < 8; ++r) { m_run[r] = -3.0e38f; l_run[r] = 0.0f; }
    v8f ctx[4] = {};

    __shared__ unsigned short p_lds[16 * 32];   // P tile, bf16

    for (int k0 = 0; k0 < SEQ; k0 += 32) {
        // ---- scores: S = Q(16x64) . K^T(64x32), two 16x16 C tiles ----
        v8f s0 = {}, s1 = {};
        #pragma unroll
        for (int t = 0; t < 2; ++t) {
            const int dk0 = t * 32;
            FragAB bk0, bk1;
            const unsigned short* p0 = Kb + (size_t)(k0 + nn) * HD + dk0 + 16 * half;
            bk0.q[0] = *(const v4i*)(p0);
            bk0.q[1] = *(const v4i*)(p0 + 8);
            const unsigned short* p1 = Kb + (size_t)(k0 + 16 + nn) * HD + dk0 + 16 * half;
            bk1.q[0] = *(const v4i*)(p1);
            bk1.q[1] = *(const v4i*)(p1 + 8);
            s0 = wmma_bf16(aQ[t], bk0, s0);
            s1 = wmma_bf16(aQ[t], bk1, s1);
        }

        // ---- scale + mask + online softmax (per-row across 16 lanes) ----
        #pragma unroll
        for (int r = 0; r < 8; ++r) {
            const int qg = q0 + r + 8 * half;
            const float mk0 = Mb[(size_t)qg * SEQ + k0 + nn];
            const float mk1 = Mb[(size_t)qg * SEQ + k0 + 16 + nn];
            float e0 = s0[r] * 0.125f - 10000.0f * (1.0f - mk0);
            float e1 = s1[r] * 0.125f - 10000.0f * (1.0f - mk1);
            float tmax = redmax16(fmaxf(e0, e1));
            const float nm = fmaxf(m_run[r], tmax);
            const float alpha = __expf(m_run[r] - nm);
            m_run[r] = nm;
            const float p0 = __expf(e0 - nm);
            const float p1 = __expf(e1 - nm);
            l_run[r] = l_run[r] * alpha + redsum16(p0 + p1);
            #pragma unroll
            for (int a = 0; a < 4; ++a) ctx[a][r] *= alpha;
            const int ql = r + 8 * half;
            p_lds[ql * 32 + nn]      = f32_bf16(p0);
            p_lds[ql * 32 + 16 + nn] = f32_bf16(p1);
        }
        __syncthreads();

        // ---- P (16x32) as A-fragment, V tiles (32x16) as B-fragments ----
        FragAB aP;
        const unsigned short* pr = p_lds + nn * 32;
    aP.q[0] = *(const v4i*)(pr + 16 * half);
    aP.q[1] = *(const v4i*)(pr + 16 * half + 8);
        #pragma unroll
        for (int a = 0; a < 4; ++a) {
            FragAB bv;
            const unsigned short* vp = Vb + (size_t)(a * 16 + nn) * SEQ + k0 + 16 * half;
            bv.q[0] = *(const v4i*)(vp);
            bv.q[1] = *(const v4i*)(vp + 8);
            ctx[a] = wmma_bf16(aP, bv, ctx[a]);
        }
        __syncthreads();

        if (k0 + 32 < SEQ)
            __builtin_prefetch(Kb + (size_t)(k0 + 32 + nn) * HD, 0, 3);
    }

    // ---- normalize and store ctx as bf16 in [B*S, HID] (row = b*S+q) ----
    #pragma unroll
    for (int r = 0; r < 8; ++r) {
        const float inv = 1.0f / l_run[r];
        const int qg = q0 + r + 8 * half;
        #pragma unroll
        for (int a = 0; a < 4; ++a) {
            const int d = a * 16 + nn;
            Ctx[(size_t)(b * SEQ + qg) * HID + h * HD + d] = f32_bf16(ctx[a][r] * inv);
        }
    }
}

// ---------------------------------------------------------------------------
// 5) Output projection + residual: out = Ctx @ Wo^T + x   (f32 out)
//    32x64 tile per wave, B fragments reused across two M-subtiles.
//    grid = (ROWS/32, HID/64), block = 32
// ---------------------------------------------------------------------------
__global__ void __launch_bounds__(32)
oproj_kernel(const unsigned short* __restrict__ Ctx,
             const unsigned short* __restrict__ Wo,
             const float* __restrict__ resid,
             float* __restrict__ out) {
    const int lane = threadIdx.x;
    const int half = lane >> 4;
    const int nn   = lane & 15;
    const int m0   = blockIdx.x * 32;
    const int n0   = blockIdx.y * 64;

    v8f acc[2][4] = {};
    const unsigned short* arow0 = Ctx + (size_t)(m0 + nn) * HID;
    const unsigned short* arow1 = Ctx + (size_t)(m0 + 16 + nn) * HID;

    for (int k0 = 0; k0 < HID; k0 += 32) {
        FragAB a0, a1;
        a0.q[0] = *(const v4i*)(arow0 + k0 + 16 * half);
        a0.q[1] = *(const v4i*)(arow0 + k0 + 16 * half + 8);
        a1.q[0] = *(const v4i*)(arow1 + k0 + 16 * half);
        a1.q[1] = *(const v4i*)(arow1 + k0 + 16 * half + 8);
        #pragma unroll
        for (int nt = 0; nt < 4; ++nt) {
            FragAB b;
            const unsigned short* wr =
                Wo + (size_t)(n0 + nt * 16 + nn) * HID + k0 + 16 * half;
            b.q[0] = *(const v4i*)(wr);
            b.q[1] = *(const v4i*)(wr + 8);
            acc[0][nt] = wmma_bf16(a0, b, acc[0][nt]);
            acc[1][nt] = wmma_bf16(a1, b, acc[1][nt]);
        }
    }

    #pragma unroll
    for (int mi = 0; mi < 2; ++mi) {
        #pragma unroll
        for (int nt = 0; nt < 4; ++nt) {
            #pragma unroll
            for (int r = 0; r < 8; ++r) {
                const int m = m0 + mi * 16 + r + 8 * half;
                const int n = n0 + nt * 16 + nn;
                out[(size_t)m * HID + n] = acc[mi][nt][r] + resid[(size_t)m * HID + n];
            }
        }
    }
}

// ---------------------------------------------------------------------------
extern "C" void kernel_launch(void* const* d_in, const int* in_sizes, int n_in,
                              void* d_out, int out_size, void* d_ws, size_t ws_size,
                              hipStream_t stream) {
    (void)in_sizes; (void)n_in; (void)out_size; (void)ws_size;
    const float* x     = (const float*)d_in[0];
    // d_in[1] = memory_tensor: unused (self-attention uses normed x as memory)
    const float* mask  = (const float*)d_in[2];
    const float* wq    = (const float*)d_in[3];
    const float* wk    = (const float*)d_in[4];
    const float* wv    = (const float*)d_in[5];
    const float* wo    = (const float*)d_in[6];
    const float* gamma = (const float*)d_in[7];
    const float* beta  = (const float*)d_in[8];
    float* out = (float*)d_out;

    char* ws = (char*)d_ws;
    unsigned short* normed = (unsigned short*)(ws);                       //  8 MB
    unsigned short* wq_b   = (unsigned short*)(ws + ((size_t)8  << 20));  //  2 MB
    unsigned short* wk_b   = (unsigned short*)(ws + ((size_t)10 << 20));
    unsigned short* wv_b   = (unsigned short*)(ws + ((size_t)12 << 20));
    unsigned short* wo_b   = (unsigned short*)(ws + ((size_t)14 << 20));
    unsigned short* Qb     = (unsigned short*)(ws + ((size_t)16 << 20));  //  8 MB
    unsigned short* Kb     = (unsigned short*)(ws + ((size_t)24 << 20));  //  8 MB
    unsigned short* Vtb    = (unsigned short*)(ws + ((size_t)32 << 20));  //  8 MB
    unsigned short* Ctxb   = (unsigned short*)(ws + ((size_t)40 << 20));  //  8 MB

    ln_bf16_kernel<<<ROWS, 256, 0, stream>>>(x, gamma, beta, normed);

    const int CN = HID * HID;
    f32_to_bf16_kernel<<<CN / 256, 256, 0, stream>>>(wq, wq_b, CN);
    f32_to_bf16_kernel<<<CN / 256, 256, 0, stream>>>(wk, wk_b, CN);
    f32_to_bf16_kernel<<<CN / 256, 256, 0, stream>>>(wv, wv_b, CN);
    f32_to_bf16_kernel<<<CN / 256, 256, 0, stream>>>(wo, wo_b, CN);

    dim3 gQKV(ROWS / 32, HID / 64, 3);
    qkv_gemm_kernel<<<gQKV, 32, 0, stream>>>(normed, wq_b, wk_b, wv_b, Qb, Kb, Vtb);

    dim3 gAtt(SEQ / 16, BATCH * NH);
    attn_kernel<<<gAtt, 32, 0, stream>>>(Qb, Kb, Vtb, mask, Ctxb);

    dim3 gO(ROWS / 32, HID / 64);
    oproj_kernel<<<gO, 32, 0, stream>>>(Ctxb, wo_b, x, out);
}
